// PointNetEncoder_78915729096885
// MI455X (gfx1250) — matrix operand order repaired and checked by reading, not compiled, not run on hardware
//
#include <hip/hip_runtime.h>

typedef _Float16 v16h __attribute__((ext_vector_type(16)));
typedef float    v8f  __attribute__((ext_vector_type(8)));

#define DEV_INLINE __device__ __forceinline__

enum { B_SZ = 8, N_PTS = 2048, M1 = 1024, M2 = 256, KNEI = 64 };

// ---------------------------------------------------------------------------
// Weight packing: f32 [Kreal][Cout] row-major -> f16 fragment-major layout.
// Element (ct, kt, lane, j) = W[kt*32 + (lane>>4)*16 + j][ct*16 + (lane&15)]
// (zero if k >= Kreal). A WMMA B-fragment is then one contiguous, 32B-aligned
// 16-half vector per lane -> 2x global_load_b128 instead of 16x b32.
// ---------------------------------------------------------------------------
__global__ __launch_bounds__(256)
void pack_weights_kernel(const float* __restrict__ W, int Kreal, int Kpad, int Cout,
                         _Float16* __restrict__ out)
{
    const int KT = Kpad >> 5;
    const int total = (Cout >> 4) * KT * 32 * 16;
    for (int idx = blockIdx.x * blockDim.x + threadIdx.x; idx < total;
         idx += gridDim.x * blockDim.x) {
        const int j    = idx & 15;
        const int lane = (idx >> 4) & 31;
        const int frag = idx >> 9;          // ct*KT + kt
        const int kt   = frag % KT;
        const int ct   = frag / KT;
        const int col  = ct * 16 + (lane & 15);
        const int k    = kt * 32 + (lane >> 4) * 16 + j;
        const float v  = (k < Kreal) ? W[k * Cout + col] : 0.f;
        out[idx] = (_Float16)v;
    }
}

// ---------------------------------------------------------------------------
// Farthest point sampling: one workgroup per cloud, fully LDS-resident.
// ---------------------------------------------------------------------------
__global__ __launch_bounds__(256)
void fps_kernel(const float* __restrict__ pos, int N, int K,
                float* __restrict__ pos_out)
{
    __shared__ float px[2048], py[2048], pz[2048], mind[2048];
    __shared__ float rbest[256];
    __shared__ int   ridx[256];
    __shared__ int   s_last;

    const int b = blockIdx.x, tid = threadIdx.x;
    const float* p = pos + (long)b * N * 3;

    for (int i = tid; i < N; i += 256) {
        px[i] = p[i * 3 + 0];
        py[i] = p[i * 3 + 1];
        pz[i] = p[i * 3 + 2];
        mind[i] = 1e10f;
    }
    if (tid == 0) {
        s_last = 0;
        pos_out[(long)b * K * 3 + 0] = p[0];
        pos_out[(long)b * K * 3 + 1] = p[1];
        pos_out[(long)b * K * 3 + 2] = p[2];
    }
    __syncthreads();

    for (int s = 1; s < K; ++s) {
        const int last = s_last;
        const float lx = px[last], ly = py[last], lz = pz[last];
        float bv = -1.f; int bi = 0;
        for (int i = tid; i < N; i += 256) {
            float dx = px[i] - lx, dy = py[i] - ly, dz = pz[i] - lz;
            float d = dx * dx + dy * dy + dz * dz;
            float m = fminf(mind[i], d);
            mind[i] = m;
            if (m > bv) { bv = m; bi = i; }   // strict > keeps smallest index
        }
        rbest[tid] = bv; ridx[tid] = bi;
        __syncthreads();
        for (int off = 128; off > 0; off >>= 1) {
            if (tid < off) {
                float ov = rbest[tid + off]; int oi = ridx[tid + off];
                if (ov > rbest[tid] || (ov == rbest[tid] && oi < ridx[tid])) {
                    rbest[tid] = ov; ridx[tid] = oi;
                }
            }
            __syncthreads();
        }
        if (tid == 0) {
            int nx = ridx[0];
            s_last = nx;
            pos_out[((long)b * K + s) * 3 + 0] = px[nx];
            pos_out[((long)b * K + s) * 3 + 1] = py[nx];
            pos_out[((long)b * K + s) * 3 + 2] = pz[nx];
        }
        __syncthreads();
    }
}

// ---------------------------------------------------------------------------
// Radius neighbor query: one wave32 per query point; ballot compaction.
// ---------------------------------------------------------------------------
__global__ __launch_bounds__(256)
void radius_kernel(const float* __restrict__ qpos, const float* __restrict__ spos,
                   int M, int N, float r2, int total,
                   int* __restrict__ nbr, int* __restrict__ cnt_out)
{
    const int wv = blockIdx.x * (blockDim.x >> 5) + (threadIdx.x >> 5);
    if (wv >= total) return;
    const int lane = threadIdx.x & 31;
    const int b = wv / M;
    const float qx = qpos[wv * 3 + 0], qy = qpos[wv * 3 + 1], qz = qpos[wv * 3 + 2];
    const float* sp = spos + (long)b * N * 3;

    int cnt = 0;
    for (int j0 = 0; j0 < N; j0 += 32) {
        const int j = j0 + lane;
        float dx = sp[j * 3 + 0] - qx;
        float dy = sp[j * 3 + 1] - qy;
        float dz = sp[j * 3 + 2] - qz;
        bool ok = (dx * dx + dy * dy + dz * dz) <= r2;
        unsigned mask = __builtin_amdgcn_ballot_w32(ok);
        int off = __popc(mask & ((1u << lane) - 1u));
        int slot = cnt + off;
        if (ok && slot < KNEI) nbr[(long)wv * KNEI + slot] = j;
        cnt += __popc(mask);
        if (cnt >= KNEI) break;
    }
    if (lane == 0) cnt_out[wv] = cnt < KNEI ? cnt : KNEI;
}

// ---------------------------------------------------------------------------
// One WMMA MLP layer over a 64-row LDS activation matrix, packed f16 weights.
// Wave w owns rows [16w, 16w+16); no cross-wave dependency between layers.
// ---------------------------------------------------------------------------
template<int KPAD, int COUT, bool FINAL>
DEV_INLINE void mlp_layer(const _Float16* __restrict__ src,
                          const _Float16* __restrict__ Wp, const float* __restrict__ Bi,
                          _Float16* __restrict__ dst, float* __restrict__ colmax,
                          int w, int lane, int cn)
{
    constexpr int KT = KPAD / 32;
    const int rlo = lane & 15;
    const int hi  = lane >> 4;
    const int rowA = w * 16 + rlo;
    for (int ct = 0; ct < COUT / 16; ++ct) {
        const int col = ct * 16 + rlo;
        const float bc = Bi[col];
        v8f acc = {0.f, 0.f, 0.f, 0.f, 0.f, 0.f, 0.f, 0.f};
        for (int kt = 0; kt < KT; ++kt) {
            v16h a;
#pragma unroll
            for (int j = 0; j < 16; ++j) {
                int ka = kt * 32 + hi * 16 + j;
                a[j] = src[rowA * KPAD + ka];
            }
            const v16h bb = *(const v16h*)(Wp + (((ct * KT) + kt) * 32 + lane) * 16);
            acc = __builtin_amdgcn_wmma_f32_16x16x32_f16(
                false, a, false, bb, (short)0, acc, false, false);
        }
        if (!FINAL) {
#pragma unroll
            for (int i = 0; i < 8; ++i) {
                int r = w * 16 + i + 8 * hi;
                float v = acc[i] + bc; v = v > 0.f ? v : 0.f;
                dst[r * COUT + col] = (_Float16)v;
            }
        } else {
            float mx = 0.f;
#pragma unroll
            for (int i = 0; i < 8; ++i) {
                int r = w * 16 + i + 8 * hi;
                float v = acc[i] + bc; v = v > 0.f ? v : 0.f;
                if (r < cn) mx = fmaxf(mx, v);
            }
            mx = fmaxf(mx, __shfl_xor(mx, 16, 32));
            if (lane < 16) colmax[w * COUT + col] = mx;
        }
    }
}

// ---------------------------------------------------------------------------
// PointConv: one 128-thread block (4 waves) per query point.
// ---------------------------------------------------------------------------
template<int FX, int FINPAD, int C1, int C2, int C3>
__global__ __launch_bounds__(128)
void pointconv_kernel(const float* __restrict__ x_src,
                      const float* __restrict__ pos_src,
                      const float* __restrict__ q_pos,
                      const int* __restrict__ nbr,
                      const int* __restrict__ cnt,
                      int M, int N,
                      const _Float16* __restrict__ W1, const float* __restrict__ B1,
                      const _Float16* __restrict__ W2, const float* __restrict__ B2,
                      const _Float16* __restrict__ W3, const float* __restrict__ B3,
                      float* __restrict__ x_out)
{
    constexpr int FIN  = FX + 3;
    constexpr int ABUF = (FINPAD > C2 ? FINPAD : C2);
    __shared__ _Float16 bufA[64 * ABUF];
    __shared__ _Float16 bufB[64 * C1];
    __shared__ float    colmax[4 * C3];

    const int q = blockIdx.x;
    const int b = q / M;
    const int tid = threadIdx.x;
    const int lane = tid & 31;
    const int w = tid >> 5;
    const int cn = cnt[q];

    const float qx = q_pos[q * 3 + 0], qy = q_pos[q * 3 + 1], qz = q_pos[q * 3 + 2];

    // gather: thread pair (2k, 2k+1) fills row k; float4 fast path for features
    {
        const int k  = tid >> 1;
        const int c0 = tid & 1;
        _Float16* row = &bufA[k * FINPAD];
        if (k < cn) {
            const int j = nbr[(long)q * KNEI + k];
            const float* xs = x_src + ((long)b * N + j) * FX;
            const float* ps = pos_src + ((long)b * N + j) * 3;
            const float rel0 = ps[0] - qx, rel1 = ps[1] - qy, rel2 = ps[2] - qz;
            for (int c4 = c0 * 4; c4 < FINPAD; c4 += 8) {
                float v0, v1, v2, v3;
                if (c4 + 4 <= FX) {
                    const float4 f4 = *(const float4*)(xs + c4);
                    v0 = f4.x; v1 = f4.y; v2 = f4.z; v3 = f4.w;
                } else {
                    float t[4];
#pragma unroll
                    for (int u = 0; u < 4; ++u) {
                        int c = c4 + u;
                        float v;
                        if (c < FX)            v = xs[c];
                        else if (c == FX)      v = rel0;
                        else if (c == FX + 1)  v = rel1;
                        else if (c == FX + 2)  v = rel2;
                        else                   v = 0.f;
                        t[u] = v;
                    }
                    v0 = t[0]; v1 = t[1]; v2 = t[2]; v3 = t[3];
                }
                row[c4 + 0] = (_Float16)v0;
                row[c4 + 1] = (_Float16)v1;
                row[c4 + 2] = (_Float16)v2;
                row[c4 + 3] = (_Float16)v3;
            }
        } else {
            for (int c4 = c0 * 4; c4 < FINPAD; c4 += 8) {
#pragma unroll
                for (int u = 0; u < 4; ++u) row[c4 + u] = (_Float16)0.f;
            }
        }
    }
    __syncthreads();

    mlp_layer<FINPAD, C1, false>(bufA, W1, B1, bufB, nullptr, w, lane, cn);
    mlp_layer<C1, C2, false>(bufB, W2, B2, bufA, nullptr, w, lane, cn);
    mlp_layer<C2, C3, true >(bufA, W3, B3, nullptr, colmax, w, lane, cn);
    __syncthreads();

    for (int c = tid; c < C3; c += 128) {
        float v = fmaxf(fmaxf(colmax[c], colmax[C3 + c]),
                        fmaxf(colmax[2 * C3 + c], colmax[3 * C3 + c]));
        x_out[(long)q * C3 + c] = v;
    }
}

// ---------------------------------------------------------------------------
// Final head layer: 32-row chunks, 4 waves; wave = (row-tile rt, col-group cg).
// ---------------------------------------------------------------------------
template<int KPAD, int COUT, bool FINAL>
DEV_INLINE void fin_layer(const _Float16* __restrict__ src,
                          const _Float16* __restrict__ Wp, const float* __restrict__ Bi,
                          _Float16* __restrict__ dst, float* __restrict__ tmpmax,
                          int rt, int cg, int lane)
{
    constexpr int KT = KPAD / 32;
    const int rlo = lane & 15;
    const int hi  = lane >> 4;
    const int rowA = rt * 16 + rlo;
    for (int ct = cg; ct < COUT / 16; ct += 2) {
        const int col = ct * 16 + rlo;
        const float bc = Bi[col];
        v8f acc = {0.f, 0.f, 0.f, 0.f, 0.f, 0.f, 0.f, 0.f};
        for (int kt = 0; kt < KT; ++kt) {
            v16h a;
#pragma unroll
            for (int j = 0; j < 16; ++j) {
                int ka = kt * 32 + hi * 16 + j;
                a[j] = src[rowA * KPAD + ka];
            }
            const v16h bb = *(const v16h*)(Wp + (((ct * KT) + kt) * 32 + lane) * 16);
            acc = __builtin_amdgcn_wmma_f32_16x16x32_f16(
                false, a, false, bb, (short)0, acc, false, false);
        }
        if (!FINAL) {
#pragma unroll
            for (int i = 0; i < 8; ++i) {
                int r = rt * 16 + i + 8 * hi;
                float v = acc[i] + bc; v = v > 0.f ? v : 0.f;
                dst[r * COUT + col] = (_Float16)v;
            }
        } else {
            float mx = 0.f;
#pragma unroll
            for (int i = 0; i < 8; ++i) {
                float v = acc[i] + bc; v = v > 0.f ? v : 0.f;
                mx = fmaxf(mx, v);
            }
            mx = fmaxf(mx, __shfl_xor(mx, 16, 32));
            if (lane < 16) tmpmax[rt * COUT + col] = mx;
        }
    }
}

// Final head: [x2|pos2] (259) -> 256 -> 128 -> 512 -> 1024, ReLU each,
// global max over 256 points. One block per batch, 8 chunks of 32 rows.
__global__ __launch_bounds__(128)
void final_mlp_kernel(const float* __restrict__ x2, const float* __restrict__ pos2,
                      const _Float16* __restrict__ W1, const float* __restrict__ B1,
                      const _Float16* __restrict__ W2, const float* __restrict__ B2,
                      const _Float16* __restrict__ W3, const float* __restrict__ B3,
                      const _Float16* __restrict__ W4, const float* __restrict__ B4,
                      float* __restrict__ out)
{
    __shared__ _Float16 X[32 * 288];   // input (288) / L2-out (128)
    __shared__ _Float16 Y[32 * 512];   // L1-out (256) / L3-out (512)
    __shared__ float    tmpmax[2 * 1024];
    __shared__ float    gmax[1024];

    const int b = blockIdx.x, tid = threadIdx.x;
    const int lane = tid & 31, w = tid >> 5;
    const int rt = w & 1, cg = w >> 1;

    for (int c = tid; c < 1024; c += 128) gmax[c] = 0.f;
    __syncthreads();

    for (int ch = 0; ch < 8; ++ch) {
        const int r0 = ch * 32;
        // gather: 4 threads per row, float4 fast path over the 256 x2 columns
        {
            const int k = tid >> 2;
            const int s = tid & 3;
            const float* xr = x2 + ((long)b * 256 + (r0 + k)) * 256;
            const float* pr = pos2 + ((long)b * 256 + (r0 + k)) * 3;
            _Float16* row = &X[k * 288];
            for (int c4 = s * 4; c4 < 288; c4 += 16) {
                float v0, v1, v2, v3;
                if (c4 + 4 <= 256) {
                    const float4 f4 = *(const float4*)(xr + c4);
                    v0 = f4.x; v1 = f4.y; v2 = f4.z; v3 = f4.w;
                } else {
                    float t[4];
#pragma unroll
                    for (int u = 0; u < 4; ++u) {
                        int c = c4 + u;
                        t[u] = (c < 259) ? pr[c - 256] : 0.f;
                    }
                    v0 = t[0]; v1 = t[1]; v2 = t[2]; v3 = t[3];
                }
                row[c4 + 0] = (_Float16)v0;
                row[c4 + 1] = (_Float16)v1;
                row[c4 + 2] = (_Float16)v2;
                row[c4 + 3] = (_Float16)v3;
            }
        }
        __syncthreads();
        fin_layer<288,  256, false>(X, W1, B1, Y, nullptr, rt, cg, lane);
        __syncthreads();
        fin_layer<256,  128, false>(Y, W2, B2, X, nullptr, rt, cg, lane);
        __syncthreads();
        fin_layer<128,  512, false>(X, W3, B3, Y, nullptr, rt, cg, lane);
        __syncthreads();
        fin_layer<512, 1024, true >(Y, W4, B4, nullptr, tmpmax, rt, cg, lane);
        __syncthreads();
        for (int c = tid; c < 1024; c += 128)
            gmax[c] = fmaxf(gmax[c], fmaxf(tmpmax[c], tmpmax[1024 + c]));
        __syncthreads();
    }
    for (int c = tid; c < 1024; c += 128) out[(long)b * 1024 + c] = gmax[c];
}

// ---------------------------------------------------------------------------
extern "C" void kernel_launch(void* const* d_in, const int* in_sizes, int n_in,
                              void* d_out, int out_size, void* d_ws, size_t ws_size,
                              hipStream_t stream)
{
    (void)in_sizes; (void)n_in; (void)out_size; (void)ws_size;

    const float* x    = (const float*)d_in[0];   // [8,2048,3]
    const float* pos  = (const float*)d_in[1];   // [8,2048,3]
    const float* p1W1 = (const float*)d_in[2];  const float* p1b1 = (const float*)d_in[3];
    const float* p1W2 = (const float*)d_in[4];  const float* p1b2 = (const float*)d_in[5];
    const float* p1W3 = (const float*)d_in[6];  const float* p1b3 = (const float*)d_in[7];
    const float* p2W1 = (const float*)d_in[8];  const float* p2b1 = (const float*)d_in[9];
    const float* p2W2 = (const float*)d_in[10]; const float* p2b2 = (const float*)d_in[11];
    const float* p2W3 = (const float*)d_in[12]; const float* p2b3 = (const float*)d_in[13];
    const float* p3W1 = (const float*)d_in[14]; const float* p3b1 = (const float*)d_in[15];
    const float* p3W2 = (const float*)d_in[16]; const float* p3b2 = (const float*)d_in[17];
    const float* p3W3 = (const float*)d_in[18]; const float* p3b3 = (const float*)d_in[19];
    const float* p3W4 = (const float*)d_in[20]; const float* p3b4 = (const float*)d_in[21];
    float* out = (float*)d_out;

    char* ws = (char*)d_ws;
    auto alloc = [&](size_t bytes) -> char* {
        char* p = ws;
        ws += (bytes + 255) & ~(size_t)255;
        return p;
    };
    float* pos1 = (float*)alloc((size_t)B_SZ * M1 * 3 * 4);
    float* pos2 = (float*)alloc((size_t)B_SZ * M2 * 3 * 4);
    float* x1   = (float*)alloc((size_t)B_SZ * M1 * 128 * 4);
    float* x2   = (float*)alloc((size_t)B_SZ * M2 * 256 * 4);
    int*   nbr1 = (int*)  alloc((size_t)B_SZ * M1 * KNEI * 4);
    int*   cnt1 = (int*)  alloc((size_t)B_SZ * M1 * 4);
    int*   nbr2 = (int*)  alloc((size_t)B_SZ * M2 * KNEI * 4);
    int*   cnt2 = (int*)  alloc((size_t)B_SZ * M2 * 4);

    // packed f16 weights (fragment-major), K zero-padded
    _Float16* q1W1 = (_Float16*)alloc((size_t) 32 *   64 * 2);
    _Float16* q1W2 = (_Float16*)alloc((size_t) 64 *   64 * 2);
    _Float16* q1W3 = (_Float16*)alloc((size_t) 64 *  128 * 2);
    _Float16* q2W1 = (_Float16*)alloc((size_t)160 *  128 * 2);
    _Float16* q2W2 = (_Float16*)alloc((size_t)128 *  128 * 2);
    _Float16* q2W3 = (_Float16*)alloc((size_t)128 *  256 * 2);
    _Float16* q3W1 = (_Float16*)alloc((size_t)288 *  256 * 2);
    _Float16* q3W2 = (_Float16*)alloc((size_t)256 *  128 * 2);
    _Float16* q3W3 = (_Float16*)alloc((size_t)128 *  512 * 2);
    _Float16* q3W4 = (_Float16*)alloc((size_t)512 * 1024 * 2);

    auto pack = [&](const float* W, int Kreal, int Kpad, int Cout, _Float16* dst) {
        int total = (Cout >> 4) * (Kpad >> 5) * 512;
        int blocks = (total + 255) / 256;
        if (blocks > 512) blocks = 512;
        pack_weights_kernel<<<blocks, 256, 0, stream>>>(W, Kreal, Kpad, Cout, dst);
    };
    pack(p1W1,   6,  32,   64, q1W1);
    pack(p1W2,  64,  64,   64, q1W2);
    pack(p1W3,  64,  64,  128, q1W3);
    pack(p2W1, 131, 160,  128, q2W1);
    pack(p2W2, 128, 128,  128, q2W2);
    pack(p2W3, 128, 128,  256, q2W3);
    pack(p3W1, 259, 288,  256, q3W1);
    pack(p3W2, 256, 256,  128, q3W2);
    pack(p3W3, 128, 128,  512, q3W3);
    pack(p3W4, 512, 512, 1024, q3W4);

    // ---- SA1: 2048 -> 1024 points, r=0.05, MLP 6->64->64->128
    fps_kernel<<<B_SZ, 256, 0, stream>>>(pos, N_PTS, M1, pos1);
    radius_kernel<<<(B_SZ * M1 + 7) / 8, 256, 0, stream>>>(
        pos1, pos, M1, N_PTS, 0.05f * 0.05f, B_SZ * M1, nbr1, cnt1);
    pointconv_kernel<3, 32, 64, 64, 128><<<B_SZ * M1, 128, 0, stream>>>(
        x, pos, pos1, nbr1, cnt1, M1, N_PTS,
        q1W1, p1b1, q1W2, p1b2, q1W3, p1b3, x1);

    // ---- SA2: 1024 -> 256 points, r=0.1, MLP 131->128->128->256
    fps_kernel<<<B_SZ, 256, 0, stream>>>(pos1, M1, M2, pos2);
    radius_kernel<<<(B_SZ * M2 + 7) / 8, 256, 0, stream>>>(
        pos2, pos1, M2, M1, 0.1f * 0.1f, B_SZ * M2, nbr2, cnt2);
    pointconv_kernel<128, 160, 128, 128, 256><<<B_SZ * M2, 128, 0, stream>>>(
        x1, pos1, pos2, nbr2, cnt2, M2, M1,
        q2W1, p2b1, q2W2, p2b2, q2W3, p2b3, x2);

    // ---- Global head: 259 -> 256 -> 128 -> 512 -> 1024, max over points
    final_mlp_kernel<<<B_SZ, 128, 0, stream>>>(
        x2, pos2, q3W1, p3b1, q3W2, p3b2, q3W3, p3b3, q3W4, p3b4, out);
}
